// EAEncoderLayer_27590869910232
// MI455X (gfx1250) — matrix operand order repaired and checked by reading, not compiled, not run on hardware
//
#include <hip/hip_runtime.h>
#include <hip/hip_bf16.h>
#include <math.h>

typedef __bf16 bf16_t;
typedef __attribute__((ext_vector_type(16))) __bf16 v16bf;
typedef __attribute__((ext_vector_type(8)))  __bf16 v8bf;
typedef __attribute__((ext_vector_type(8)))  float   v8f;

enum { EPI_BIAS = 1, EPI_GELU = 2, EPI_RES = 4, EPI_BF16 = 8, EPI_TRANS = 16 };

#define BM 128
#define BN 128
#define BK 32

static __device__ __forceinline__ v8f wmma_bf16(v16bf a, v16bf b, v8f c) {
  // D = A(16x32 bf16) * B(32x16 bf16) + C(16x16 f32)
  return __builtin_amdgcn_wmma_f32_16x16x32_bf16(false, a, false, b, (short)0, c,
                                                 false, false);
}

static __device__ __forceinline__ v16bf concat8(v8bf lo, v8bf hi) {
  return __builtin_shufflevector(lo, hi, 0, 1, 2, 3, 4, 5, 6, 7, 8, 9, 10, 11, 12, 13,
                                 14, 15);
}

// A fragment: 16x32 bf16 tile, row-major in LDS, leading dim ld (elements).
static __device__ __forceinline__ v16bf load_frag_a(const bf16_t* p, int ld) {
  const int lane = threadIdx.x & 31;
  const int half = lane >> 4;
  const bf16_t* rp = p + (lane & 15) * ld + half * 8;
  v8bf lo = *reinterpret_cast<const v8bf*>(rp);
  v8bf hi = *reinterpret_cast<const v8bf*>(rp + 16);
  return concat8(lo, hi);
}

// B fragment from N-major (transposed) storage: B(k,n) = p[(n0+n)*ld + k0 + k].
static __device__ __forceinline__ v16bf load_frag_bT(const bf16_t* p, int ld, int n0,
                                                     int k0) {
  const int lane = threadIdx.x & 31;
  const int half = lane >> 4;
  const bf16_t* rp = p + (long)(n0 + (lane & 15)) * ld + k0 + half * 8;
  v8bf lo = *reinterpret_cast<const v8bf*>(rp);
  v8bf hi = *reinterpret_cast<const v8bf*>(rp + 16);
  return concat8(lo, hi);
}

// Async global->LDS 16B transfer (CDNA5 GLOBAL_LOAD_ASYNC_TO_LDS_B128, ASYNCcnt).
static __device__ __forceinline__ void async_copy_b128(unsigned ldsoff,
                                                       const void* gaddr) {
  asm volatile("global_load_async_to_lds_b128 %0, %1, off"
               :
               : "v"(ldsoff), "v"((unsigned long long)(uintptr_t)gaddr)
               : "memory");
}
static __device__ __forceinline__ void async_wait0() {
  asm volatile("s_wait_asynccnt 0" ::: "memory");
}

// Lane-XOR butterfly exchange via DS_SWIZZLE_B32 (xor_mask in offset[14:10]).
#define SWZ_XOR_F(x, imm) \
  __int_as_float(__builtin_amdgcn_ds_swizzle(__float_as_int(x), (imm)))
static __device__ __forceinline__ float halfmax16(float x) {
  x = fmaxf(x, SWZ_XOR_F(x, 0x201f));  // xor 8
  x = fmaxf(x, SWZ_XOR_F(x, 0x101f));  // xor 4
  x = fmaxf(x, SWZ_XOR_F(x, 0x081f));  // xor 2
  x = fmaxf(x, SWZ_XOR_F(x, 0x041f));  // xor 1
  return x;
}
static __device__ __forceinline__ float halfsum16(float x) {
  x += SWZ_XOR_F(x, 0x201f);
  x += SWZ_XOR_F(x, 0x101f);
  x += SWZ_XOR_F(x, 0x081f);
  x += SWZ_XOR_F(x, 0x041f);
  return x;
}

// ---- fp32 -> bf16 conversion (row-major copy) -------------------------------
__global__ void cvt_f32_to_bf16_kernel(const float* __restrict__ s,
                                       bf16_t* __restrict__ d, long n) {
  long i0 = ((long)blockIdx.x * blockDim.x + threadIdx.x) * 4;
  if (i0 + 4 <= n) {
    float4 v = *reinterpret_cast<const float4*>(s + i0);
    d[i0 + 0] = (bf16_t)v.x;
    d[i0 + 1] = (bf16_t)v.y;
    d[i0 + 2] = (bf16_t)v.z;
    d[i0 + 3] = (bf16_t)v.w;
  } else {
    for (long i = i0; i < n; ++i) d[i] = (bf16_t)s[i];
  }
}

// ---- fp32 (K x N) -> bf16 transposed (N x K), batched over blockIdx.z -------
__global__ __launch_bounds__(256) void cvt_f32_to_bf16_T_kernel(
    const float* __restrict__ in, bf16_t* __restrict__ out, int K, int N) {
  __shared__ float tile[32][33];
  const long base = (long)blockIdx.z * K * N;
  const int k0 = blockIdx.y * 32, n0 = blockIdx.x * 32;
  const int tx = threadIdx.x & 31, ty = threadIdx.x >> 5;  // 32 x 8
  for (int i = ty; i < 32; i += 8)
    if (k0 + i < K && n0 + tx < N)
      tile[i][tx] = in[base + (long)(k0 + i) * N + n0 + tx];
  __syncthreads();
  for (int i = ty; i < 32; i += 8)
    if (n0 + i < N && k0 + tx < K)
      out[base + (long)(n0 + i) * K + k0 + tx] = (bf16_t)tile[tx][i];
}

// ---- Generic batched bf16 GEMM with fused epilogue (compile-time EPI) -------
// C[z] = A[z] (Mr x K, row-major) * B[z] (K x N, stored N-major/transposed, ldb=K)
// ALIGNED: dims are tile multiples -> async double-buffered pipeline, no guards.
template <int EPI, bool ALIGNED>
__global__ __launch_bounds__(256) void gemm_bf16_kernel(
    const bf16_t* __restrict__ A, const bf16_t* __restrict__ Bm,
    const float* __restrict__ bias, const float* __restrict__ Res,
    float* __restrict__ Cf, bf16_t* __restrict__ Cb, int Mr, int N, int K, int zdiv,
    long sA1, long sA2, long sB1, long sB2, long sC1, long sC2, int lda, int ldb,
    int ldc) {
  __shared__ __align__(16) bf16_t As[2][BM][BK];  // 16 KB (double-buffered)
  __shared__ __align__(16) bf16_t Bt[2][BN][BK];  // 16 KB (N-major)

  const int z = blockIdx.z;
  const long offA = (long)(z / zdiv) * sA1 + (long)(z % zdiv) * sA2;
  const long offB = (long)(z / zdiv) * sB1 + (long)(z % zdiv) * sB2;
  const long offC = (long)(z / zdiv) * sC1 + (long)(z % zdiv) * sC2;
  const bf16_t* Ap = A + offA;
  const bf16_t* Bp = Bm + offB;

  const int bm = blockIdx.y * BM;
  const int bn = blockIdx.x * BN;
  const int tid = threadIdx.x;
  const int wave = tid >> 5;
  const int lane = tid & 31;
  const int half = lane >> 4;
  const int coln = lane & 15;

  v8f acc[8] = {};

  if constexpr (ALIGNED) {
    // Per-thread 16B chunk coords: rows r0 and r0+64, fixed column segment.
    const int r0 = tid >> 2, cc0 = (tid & 3) * 8;
    const bf16_t* gA0 = Ap + (long)(bm + r0) * lda + cc0;
    const bf16_t* gA1 = Ap + (long)(bm + r0 + 64) * lda + cc0;
    const bf16_t* gB0 = Bp + (long)(bn + r0) * ldb + cc0;
    const bf16_t* gB1 = Bp + (long)(bn + r0 + 64) * ldb + cc0;
    auto stage = [&](int buf, int k0) {
      async_copy_b128((unsigned)(uintptr_t)&As[buf][r0][cc0], gA0 + k0);
      async_copy_b128((unsigned)(uintptr_t)&As[buf][r0 + 64][cc0], gA1 + k0);
      async_copy_b128((unsigned)(uintptr_t)&Bt[buf][r0][cc0], gB0 + k0);
      async_copy_b128((unsigned)(uintptr_t)&Bt[buf][r0 + 64][cc0], gB1 + k0);
    };
    stage(0, 0);
    async_wait0();
    __syncthreads();
    const int nsteps = K / BK;
    for (int s = 0; s < nsteps; ++s) {
      const int buf = s & 1;
      if (s + 1 < nsteps) stage(1 - buf, (s + 1) * BK);  // prefetch next tile
      v16bf af = load_frag_a(&As[buf][wave * 16][0], BK);
#pragma unroll
      for (int ct = 0; ct < 8; ++ct) {
        v16bf bfg = load_frag_bT(&Bt[buf][0][0], BK, ct * 16, 0);
        acc[ct] = wmma_bf16(af, bfg, acc[ct]);
      }
      async_wait0();     // next tile landed in the other buffer
      __syncthreads();   // all waves done reading this buffer
    }
  } else {
    for (int k0 = 0; k0 < K; k0 += BK) {
      for (int c = tid; c < (BM * BK) / 8; c += 256) {
        int r = c >> 2, cc = (c & 3) * 8;
        int gr = bm + r, gc = k0 + cc;
        if (gr < Mr && gc + 8 <= K) {
          *reinterpret_cast<uint4*>(&As[0][r][cc]) =
              *reinterpret_cast<const uint4*>(Ap + (long)gr * lda + gc);
        } else {
          for (int j = 0; j < 8; ++j)
            As[0][r][cc + j] =
                (gr < Mr && gc + j < K) ? Ap[(long)gr * lda + gc + j] : (bf16_t)0.f;
        }
      }
      for (int c = tid; c < (BN * BK) / 8; c += 256) {
        int r = c >> 2, cc = (c & 3) * 8;
        int gn = bn + r, gk = k0 + cc;
        if (gn < N && gk + 8 <= K) {
          *reinterpret_cast<uint4*>(&Bt[0][r][cc]) =
              *reinterpret_cast<const uint4*>(Bp + (long)gn * ldb + gk);
        } else {
          for (int j = 0; j < 8; ++j)
            Bt[0][r][cc + j] =
                (gn < N && gk + j < K) ? Bp[(long)gn * ldb + gk + j] : (bf16_t)0.f;
        }
      }
      __syncthreads();
      v16bf af = load_frag_a(&As[0][wave * 16][0], BK);
#pragma unroll
      for (int ct = 0; ct < 8; ++ct) {
        v16bf bfg = load_frag_bT(&Bt[0][0][0], BK, ct * 16, 0);
        acc[ct] = wmma_bf16(af, bfg, acc[ct]);
      }
      __syncthreads();
    }
  }

  // Epilogue: C layout row = vgpr + 8*half, col = lane%16
  const int rbase = bm + wave * 16 + half * 8;
#pragma unroll
  for (int ct = 0; ct < 8; ++ct) {
    int col = bn + ct * 16 + coln;
    if (!ALIGNED && col >= N) continue;
#pragma unroll
    for (int v = 0; v < 8; ++v) {
      int row = rbase + v;
      if (!ALIGNED && row >= Mr) continue;
      float val = acc[ct][v];
      if constexpr (EPI & EPI_BIAS) val += bias[col];
      if constexpr (EPI & EPI_GELU)
        val = 0.5f * val * (1.f + erff(val * 0.70710678118f));
      if constexpr (EPI & EPI_RES) val += Res[offC + (long)row * ldc + col];
      long idx;
      if constexpr (EPI & EPI_TRANS)
        idx = offC + (long)col * ldc + row;
      else
        idx = offC + (long)row * ldc + col;
      if constexpr (EPI & EPI_BF16)
        Cb[idx] = (bf16_t)val;
      else
        Cf[idx] = val;
    }
  }
}

// ---- Flash attention over memory-projected scores ---------------------------
// attn = softmax(q_mem k_mem^T / sqrt(dh)) ; ctx = attn @ V_head
// Vt is the transposed V: layout (D, B*S).
#define ATT_QB 128
#define ATT_KB 32
__global__ __launch_bounds__(256) void flash_attn_kernel(
    const bf16_t* __restrict__ qmem, const bf16_t* __restrict__ kmem,
    const bf16_t* __restrict__ Vt, bf16_t* __restrict__ ctxb, int S, int D, int dh,
    int Mdim, int Hn, long BSld) {
  __shared__ __align__(16) bf16_t Qs[ATT_QB][64];       // 16 KB
  __shared__ __align__(16) bf16_t Ks[2][ATT_KB][64];    // 8 KB (keys x mem-dim)
  __shared__ __align__(16) bf16_t Vts[2][64][ATT_KB];   // 8 KB (dh x keys, N-major)
  __shared__ __align__(16) bf16_t Ps[8][16][32];        // 8 KB (per-wave P staging)

  const int z = blockIdx.y;
  const int b = z / Hn, h = z % Hn;
  const int q0 = blockIdx.x * ATT_QB;
  const int tid = threadIdx.x, wave = tid >> 5, lane = tid & 31;
  const int half = lane >> 4, coln = lane & 15;

  const bf16_t* Qp = qmem + (long)z * S * Mdim;
  const bf16_t* Kp = kmem + (long)z * S * Mdim;
  const bf16_t* Vp = Vt + (long)(h * dh) * BSld + (long)b * S;

  for (int c = tid; c < (ATT_QB * 64) / 8; c += 256) {
    int r = c >> 3, cc = (c & 7) * 8;
    *reinterpret_cast<uint4*>(&Qs[r][cc]) =
        *reinterpret_cast<const uint4*>(Qp + (long)(q0 + r) * Mdim + cc);
  }
  __syncthreads();

  v16bf qf0 = load_frag_a(&Qs[wave * 16][0], 64);       // contraction dims 0..31
  v16bf qf1 = load_frag_a(&Qs[wave * 16][0] + 32, 64);  // contraction dims 32..63

  // Per-thread async-copy targets
  const int kr = tid >> 3, kc = (tid & 7) * 8;   // K tile: 32 x 64
  const int vr = tid >> 2, vc = (tid & 3) * 8;   // V^T tile: 64 x 32
  auto stageKV = [&](int buf, int kt) {
    async_copy_b128((unsigned)(uintptr_t)(&Ks[buf][kr][kc]),
                    Kp + (long)(kt + kr) * Mdim + kc);
    async_copy_b128((unsigned)(uintptr_t)(&Vts[buf][vr][vc]),
                    Vp + (long)vr * BSld + kt + vc);
  };

  float mrun[8], lrun[8];
  v8f o[4] = {};
#pragma unroll
  for (int v = 0; v < 8; ++v) { mrun[v] = -3.0e38f; lrun[v] = 0.f; }
  const float scale = 0.125f;  // 1/sqrt(dh), dh = 64

  stageKV(0, 0);
  async_wait0();
  __syncthreads();

  for (int kt = 0, step = 0; kt < S; kt += ATT_KB, ++step) {
    const int buf = step & 1;
    if (kt + ATT_KB < S) stageKV(1 - buf, kt + ATT_KB);  // prefetch next K/V tiles

    // scores: 16 queries x 32 keys, contraction over Mdim=64 (two WMMA K-chunks)
    v8f s0 = {}, s1 = {};
    s0 = wmma_bf16(qf0, load_frag_bT(&Ks[buf][0][0], 64, 0, 0), s0);
    s0 = wmma_bf16(qf1, load_frag_bT(&Ks[buf][0][0], 64, 0, 32), s0);
    s1 = wmma_bf16(qf0, load_frag_bT(&Ks[buf][0][0], 64, 16, 0), s1);
    s1 = wmma_bf16(qf1, load_frag_bT(&Ks[buf][0][0], 64, 16, 32), s1);

#pragma unroll
    for (int v = 0; v < 8; ++v) {
      float a = s0[v] * scale;
      float c2 = s1[v] * scale;
      float rm = halfmax16(fmaxf(a, c2));
      float nm = fmaxf(mrun[v], rm);
      float corr = __expf(mrun[v] - nm);
      float p0 = __expf(a - nm);
      float p1 = __expf(c2 - nm);
      float ps = halfsum16(p0 + p1);
      lrun[v] = lrun[v] * corr + ps;
      mrun[v] = nm;
#pragma unroll
      for (int j = 0; j < 4; ++j) o[j][v] *= corr;
      int prow = v + half * 8;
      Ps[wave][prow][coln] = (bf16_t)p0;
      Ps[wave][prow][16 + coln] = (bf16_t)p1;
    }
    // per-wave LDS RAW: DS ops are in-order per wave; wait before re-reading as A-frag
    asm volatile("s_wait_dscnt 0" ::: "memory");
    v16bf pf = load_frag_a(&Ps[wave][0][0], 32);
#pragma unroll
    for (int j = 0; j < 4; ++j)
      o[j] = wmma_bf16(pf, load_frag_bT(&Vts[buf][0][0], ATT_KB, j * 16, 0), o[j]);

    async_wait0();     // next tiles landed in the other buffers
    __syncthreads();   // all waves done reading this buffer
  }

  const int srow = q0 + wave * 16 + half * 8;
  bf16_t* Cp = ctxb + (long)b * S * D + h * dh;
#pragma unroll
  for (int j = 0; j < 4; ++j) {
    int col = j * 16 + coln;
#pragma unroll
    for (int v = 0; v < 8; ++v) {
      float denom = lrun[v];
      float val = (denom > 0.f) ? o[j][v] / denom : 0.f;
      Cp[(long)(srow + v) * D + col] = (bf16_t)val;
    }
  }
}

// ---- LayerNorm (fp32 in; fp32 out + optional bf16 mirror) -------------------
__global__ __launch_bounds__(256) void layernorm_kernel(
    const float* __restrict__ X, const float* __restrict__ g,
    const float* __restrict__ bta, float* __restrict__ Yf, bf16_t* __restrict__ Yb,
    int Ddim) {
  __shared__ float red[512];
  const long row = blockIdx.x;
  const float* xr = X + row * Ddim;
  float s = 0.f, s2 = 0.f;
  for (int i = threadIdx.x; i < Ddim; i += blockDim.x) {
    float v = xr[i];
    s += v;
    s2 += v * v;
  }
  red[threadIdx.x] = s;
  red[256 + threadIdx.x] = s2;
  __syncthreads();
  for (int off = 128; off > 0; off >>= 1) {
    if ((int)threadIdx.x < off) {
      red[threadIdx.x] += red[threadIdx.x + off];
      red[256 + threadIdx.x] += red[256 + threadIdx.x + off];
    }
    __syncthreads();
  }
  float mean = red[0] / Ddim;
  float var = red[256] / Ddim - mean * mean;
  float inv = rsqrtf(var + 1e-5f);
  for (int i = threadIdx.x; i < Ddim; i += blockDim.x) {
    float y = (xr[i] - mean) * inv * g[i] + bta[i];
    if (Yf) Yf[row * Ddim + i] = y;
    if (Yb) Yb[row * Ddim + i] = (bf16_t)y;
  }
}

// ---- Host orchestration -----------------------------------------------------
extern "C" void kernel_launch(void* const* d_in, const int* in_sizes, int n_in,
                              void* d_out, int out_size, void* d_ws, size_t ws_size,
                              hipStream_t stream) {
  (void)in_sizes; (void)n_in; (void)out_size; (void)ws_size;
  constexpr int B = 2, S = 2048, D = 1024, Hh = 16, dh = 64, Mm = 64, F = 4096;
  const long BS = (long)B * S;

  const float* x = (const float*)d_in[0];
  const float* Wq = (const float*)d_in[1];
  const float* Wk = (const float*)d_in[2];
  const float* Wv = (const float*)d_in[3];
  const float* Wo = (const float*)d_in[4];
  const float* Mk = (const float*)d_in[5];
  const float* ln1g = (const float*)d_in[6];
  const float* ln1b = (const float*)d_in[7];
  const float* ln2g = (const float*)d_in[8];
  const float* ln2b = (const float*)d_in[9];
  const float* W1 = (const float*)d_in[10];
  const float* b1 = (const float*)d_in[11];
  const float* W2 = (const float*)d_in[12];
  const float* b2 = (const float*)d_in[13];

  char* wsp = (char*)d_ws;
  auto alloc = [&](size_t bytes) -> char* {
    char* p = wsp;
    wsp += (bytes + 255) & ~(size_t)255;
    return p;
  };
  bf16_t* xb  = (bf16_t*)alloc(BS * D * 2);
  bf16_t* Wqb = (bf16_t*)alloc((size_t)D * D * 2);   // transposed (N,K)
  bf16_t* Wkb = (bf16_t*)alloc((size_t)D * D * 2);
  bf16_t* Wvb = (bf16_t*)alloc((size_t)D * D * 2);
  bf16_t* Wob = (bf16_t*)alloc((size_t)D * D * 2);
  bf16_t* Mkb = (bf16_t*)alloc((size_t)Hh * dh * Mm * 2);  // per-head (M,dh)
  bf16_t* W1b = (bf16_t*)alloc((size_t)D * F * 2);   // (F,D)
  bf16_t* W2b = (bf16_t*)alloc((size_t)F * D * 2);   // (D,F)
  bf16_t* Qb  = (bf16_t*)alloc(BS * D * 2);
  bf16_t* Kb  = (bf16_t*)alloc(BS * D * 2);
  bf16_t* Vtb = (bf16_t*)alloc(BS * D * 2);          // transposed V: (D, BS)
  bf16_t* qm  = (bf16_t*)alloc((size_t)B * Hh * S * Mm * 2);
  bf16_t* km  = (bf16_t*)alloc((size_t)B * Hh * S * Mm * 2);
  bf16_t* ctxb = (bf16_t*)alloc(BS * D * 2);
  float*  t1  = (float*)alloc(BS * D * 4);
  float*  x1f = (float*)alloc(BS * D * 4);
  bf16_t* x1b = (bf16_t*)alloc(BS * D * 2);
  bf16_t* h1b = (bf16_t*)alloc(BS * F * 2);
  float*  t2  = (float*)alloc(BS * D * 4);

  auto cvt = [&](const float* s, bf16_t* dp, long n) {
    long thr = (n + 3) / 4;
    cvt_f32_to_bf16_kernel<<<dim3((unsigned)((thr + 255) / 256)), dim3(256), 0,
                             stream>>>(s, dp, n);
  };
  auto cvtT = [&](const float* s, bf16_t* dp, int K, int N, int batch) {
    dim3 g((N + 31) / 32, (K + 31) / 32, batch);
    cvt_f32_to_bf16_T_kernel<<<g, dim3(256), 0, stream>>>(s, dp, K, N);
  };
  cvt(x, xb, BS * D);
  cvtT(Wq, Wqb, D, D, 1);
  cvtT(Wk, Wkb, D, D, 1);
  cvtT(Wv, Wvb, D, D, 1);
  cvtT(Wo, Wob, D, D, 1);
  cvtT(Mk, Mkb, dh, Mm, Hh);
  cvtT(W1, W1b, D, F, 1);
  cvtT(W2, W2b, F, D, 1);

  const dim3 blk(256);
  // Q/K projections -> row-major bf16; V projection -> transposed (D, BS) bf16
  {
    dim3 g(D / BN, (unsigned)(BS / BM));
    gemm_bf16_kernel<EPI_BF16, true><<<g, blk, 0, stream>>>(
        xb, Wqb, nullptr, nullptr, nullptr, Qb, (int)BS, D, D, 1, 0, 0, 0, 0, 0, 0, D,
        D, D);
    gemm_bf16_kernel<EPI_BF16, true><<<g, blk, 0, stream>>>(
        xb, Wkb, nullptr, nullptr, nullptr, Kb, (int)BS, D, D, 1, 0, 0, 0, 0, 0, 0, D,
        D, D);
    gemm_bf16_kernel<EPI_BF16 | EPI_TRANS, true><<<g, blk, 0, stream>>>(
        xb, Wvb, nullptr, nullptr, nullptr, Vtb, (int)BS, D, D, 1, 0, 0, 0, 0, 0, 0, D,
        D, (int)BS);
  }
  // Memory-key projections: per (b,h): (S x dh) @ (dh x M) -> (B,H,S,M); N=64 -> guarded
  {
    dim3 g((Mm + BN - 1) / BN, S / BM, B * Hh);
    gemm_bf16_kernel<EPI_BF16, false><<<g, blk, 0, stream>>>(
        Qb, Mkb, nullptr, nullptr, nullptr, qm, S, Mm, dh, Hh, (long)S * D, dh, 0,
        (long)dh * Mm, (long)Hh * S * Mm, (long)S * Mm, D, dh, Mm);
    gemm_bf16_kernel<EPI_BF16, false><<<g, blk, 0, stream>>>(
        Kb, Mkb, nullptr, nullptr, nullptr, km, S, Mm, dh, Hh, (long)S * D, dh, 0,
        (long)dh * Mm, (long)Hh * S * Mm, (long)S * Mm, D, dh, Mm);
  }
  // Flash attention -> ctx (B,S,D) bf16
  flash_attn_kernel<<<dim3(S / ATT_QB, B * Hh), blk, 0, stream>>>(qm, km, Vtb, ctxb, S,
                                                                  D, dh, Mm, Hh, BS);
  // Output projection + residual -> fp32, then LN1 (fp32 + bf16 mirror)
  {
    dim3 g(D / BN, (unsigned)(BS / BM));
    gemm_bf16_kernel<EPI_RES, true><<<g, blk, 0, stream>>>(
        ctxb, Wob, nullptr, x, t1, nullptr, (int)BS, D, D, 1, 0, 0, 0, 0, 0, 0, D, D,
        D);
  }
  layernorm_kernel<<<dim3((unsigned)BS), blk, 0, stream>>>(t1, ln1g, ln1b, x1f, x1b,
                                                           D);
  // FFN: gelu(x1 @ W1 + b1) @ W2 + b2 + x1, then LN2 -> d_out (fp32)
  {
    dim3 g1(F / BN, (unsigned)(BS / BM));
    gemm_bf16_kernel<EPI_BIAS | EPI_GELU | EPI_BF16, true><<<g1, blk, 0, stream>>>(
        x1b, W1b, b1, nullptr, nullptr, h1b, (int)BS, F, D, 1, 0, 0, 0, 0, 0, 0, D, D,
        F);
    dim3 g2(D / BN, (unsigned)(BS / BM));
    gemm_bf16_kernel<EPI_BIAS | EPI_RES, true><<<g2, blk, 0, stream>>>(
        h1b, W2b, b2, x1f, t2, nullptr, (int)BS, D, F, 1, 0, 0, 0, 0, 0, 0, F, F, D);
  }
  layernorm_kernel<<<dim3((unsigned)BS), blk, 0, stream>>>(t2, ln2g, ln2b,
                                                           (float*)d_out, nullptr, D);
}
